// Contrastive_loss_10462540333089
// MI455X (gfx1250) — matrix operand order repaired and checked by reading, not compiled, not run on hardware
//
#include <hip/hip_runtime.h>
#include <hip/hip_bf16.h>
#include <stdint.h>

// Contrastive loss (GRACE-style) for N=16384, D=128, tau=0.5 on MI455X.
// Fused: normalize -> bf16x3 split GEMM via v_wmma_f32_16x16x32_bf16 ->
// exp -> row/col/diag reductions -> final log-mean. NxN matrix never hits HBM.

typedef __attribute__((ext_vector_type(16))) __bf16 v16bf;
typedef __attribute__((ext_vector_type(8)))  float  v8f;

union Frag { uint4 q[2]; v16bf v; };

__device__ __forceinline__ unsigned short f2bf_rne(float f) {
  unsigned int u = __float_as_uint(f);
  u += 0x7FFFu + ((u >> 16) & 1u);          // round-to-nearest-even
  return (unsigned short)(u >> 16);
}
__device__ __forceinline__ float bf2f(unsigned short h) {
  return __uint_as_float(((unsigned int)h) << 16);
}

// ---------------------------------------------------------------- init ----
__global__ void init_zero_kernel(float* __restrict__ base, int count,
                                 float* __restrict__ out) {
  int i = blockIdx.x * blockDim.x + threadIdx.x;
  if (i < count) base[i] = 0.0f;
  if (i == 0) out[0] = 0.0f;
}

// -------------------------------------------- normalize + bf16 hi/lo split
// One wave (32 lanes) per row; 4 floats per lane. warpSize==32 on gfx1250.
__global__ __launch_bounds__(256)
void normalize_split_kernel(const float* __restrict__ z1,
                            const float* __restrict__ z2,
                            unsigned short* __restrict__ a_hi,
                            unsigned short* __restrict__ a_lo,
                            unsigned short* __restrict__ b_hi,
                            unsigned short* __restrict__ b_lo,
                            int N) {
  const int lane = threadIdx.x & 31;
  const int wave = threadIdx.x >> 5;
  const int r = blockIdx.x * 8 + wave;
  if (r >= 2 * N) return;

  const float* src;
  unsigned short *dh, *dl;
  if (r < N) {
    size_t rr = (size_t)r;
    src = z1 + rr * 128; dh = a_hi + rr * 128; dl = a_lo + rr * 128;
  } else {
    size_t rr = (size_t)(r - N);
    src = z2 + rr * 128; dh = b_hi + rr * 128; dl = b_lo + rr * 128;
  }

  float4 x = ((const float4*)src)[lane];
  float s = x.x * x.x + x.y * x.y + x.z * x.z + x.w * x.w;
  s += __shfl_xor(s, 1, 32);
  s += __shfl_xor(s, 2, 32);
  s += __shfl_xor(s, 4, 32);
  s += __shfl_xor(s, 8, 32);
  s += __shfl_xor(s, 16, 32);
  const float inv = 1.0f / fmaxf(sqrtf(s), 1e-12f);   // F.normalize eps

  float y[4] = {x.x * inv, x.y * inv, x.z * inv, x.w * inv};
  unsigned short h[4], l[4];
#pragma unroll
  for (int i = 0; i < 4; ++i) {
    h[i] = f2bf_rne(y[i]);
    l[i] = f2bf_rne(y[i] - bf2f(h[i]));   // residual for split-precision GEMM
  }
  uint2 ph, pl;
  ph.x = (unsigned)h[0] | ((unsigned)h[1] << 16);
  ph.y = (unsigned)h[2] | ((unsigned)h[3] << 16);
  pl.x = (unsigned)l[0] | ((unsigned)l[1] << 16);
  pl.y = (unsigned)l[2] | ((unsigned)l[3] << 16);
  ((uint2*)dh)[lane] = ph;
  ((uint2*)dl)[lane] = pl;
}

// ------------------------------------- fused GEMM + exp + row/col/diag ----
// Block: 128 (M) x 64 (N) output region, 256 threads = 8 waves (2x4 over NxM).
// Each wave: 2x2 tiles of 16x16. K=128 staged in two 64-wide LDS chunks,
// rows padded to 72 shorts (144B) -> conflict-free ds_load_b128 fragments.
__global__ __launch_bounds__(256)
void sim_exp_reduce_kernel(const unsigned short* __restrict__ a_hi,
                           const unsigned short* __restrict__ a_lo,
                           const unsigned short* __restrict__ b_hi,
                           const unsigned short* __restrict__ b_lo,
                           float* __restrict__ rowsum,
                           float* __restrict__ colsum,
                           float* __restrict__ diag,
                           int N) {
  __shared__ __align__(16) unsigned short As_hi[128][72];
  __shared__ __align__(16) unsigned short As_lo[128][72];
  __shared__ __align__(16) unsigned short Bs_hi[64][72];
  __shared__ __align__(16) unsigned short Bs_lo[64][72];
  __shared__ float red_row[128];
  __shared__ float red_col[64];

  const int tid   = threadIdx.x;
  const int lane  = tid & 31;
  const int lr    = lane & 15;   // col (B/C) or row (A) selector
  const int lh    = lane >> 4;   // half-wave selector
  const int wave  = tid >> 5;
  const int waveM = wave >> 1;   // 0..3 : 32-row slice of the 128-row block
  const int waveN = wave & 1;    // 0..1 : 32-col slice of the 64-col block

  const int m0 = blockIdx.y * 128;
  const int n0 = blockIdx.x * 64;

  if (tid < 128) red_row[tid] = 0.0f;
  if (tid < 64)  red_col[tid] = 0.0f;

  const v8f vzero = {0.f, 0.f, 0.f, 0.f, 0.f, 0.f, 0.f, 0.f};
  v8f acc[2][2];
  acc[0][0] = vzero; acc[0][1] = vzero; acc[1][0] = vzero; acc[1][1] = vzero;

  for (int kc = 0; kc < 2; ++kc) {          // two K=64 chunks over D=128
    // Stage A (128x64) and B (64x64), hi+lo, in 16-byte granules.
    for (int g = tid; g < 1024; g += 256) {
      const int row = g >> 3, c = g & 7;
      const size_t goff = (size_t)(m0 + row) * 128 + (size_t)kc * 64;
      *(uint4*)&As_hi[row][c * 8] = ((const uint4*)(a_hi + goff))[c];
      *(uint4*)&As_lo[row][c * 8] = ((const uint4*)(a_lo + goff))[c];
    }
    for (int g = tid; g < 512; g += 256) {
      const int row = g >> 3, c = g & 7;
      const size_t goff = (size_t)(n0 + row) * 128 + (size_t)kc * 64;
      *(uint4*)&Bs_hi[row][c * 8] = ((const uint4*)(b_hi + goff))[c];
      *(uint4*)&Bs_lo[row][c * 8] = ((const uint4*)(b_lo + goff))[c];
    }
    __syncthreads();

#pragma unroll
    for (int ks = 0; ks < 2; ++ks) {        // two K=32 WMMA steps per chunk
      const int kb = ks * 32;
      Frag ah[2], al[2], bh[2], bl[2];
#pragma unroll
      for (int t = 0; t < 2; ++t) {
        // A fragment (16x32, ISA 7.12.2): lane holds row lr;
        // elems 0-7 -> K = lh*8..+7, elems 8-15 -> K = 16+lh*8..+7.
        const int ar = waveM * 32 + t * 16 + lr;
        ah[t].q[0] = *(const uint4*)&As_hi[ar][kb + lh * 8];
        ah[t].q[1] = *(const uint4*)&As_hi[ar][kb + 16 + lh * 8];
        al[t].q[0] = *(const uint4*)&As_lo[ar][kb + lh * 8];
        al[t].q[1] = *(const uint4*)&As_lo[ar][kb + 16 + lh * 8];
        // B fragment (32x16): lane holds col lr; elems 0-7 -> K = lh*8..+7, elems 8-15 -> K = 16+lh*8..+7.
        const int br = waveN * 32 + t * 16 + lr;
        bh[t].q[0] = *(const uint4*)&Bs_hi[br][kb + lh * 8];
        bh[t].q[1] = *(const uint4*)&Bs_hi[br][kb + 16 + lh * 8];
        bl[t].q[0] = *(const uint4*)&Bs_lo[br][kb + lh * 8];
        bl[t].q[1] = *(const uint4*)&Bs_lo[br][kb + 16 + lh * 8];
      }
#pragma unroll
      for (int tm = 0; tm < 2; ++tm)
#pragma unroll
        for (int tn = 0; tn < 2; ++tn) {
          // bf16x3 split: ah*bh + ah*bl + al*bh  (al*bl ~2^-16, dropped)
          acc[tm][tn] = __builtin_amdgcn_wmma_f32_16x16x32_bf16(
              false, ah[tm].v, false, bh[tn].v, (short)0, acc[tm][tn], false, false);
          acc[tm][tn] = __builtin_amdgcn_wmma_f32_16x16x32_bf16(
              false, ah[tm].v, false, bl[tn].v, (short)0, acc[tm][tn], false, false);
          acc[tm][tn] = __builtin_amdgcn_wmma_f32_16x16x32_bf16(
              false, al[tm].v, false, bh[tn].v, (short)0, acc[tm][tn], false, false);
        }
    }
    __syncthreads();
  }

  // Epilogue: S = exp(2*dot); diag store; row/col partial sums.
  // C/D layout: VGPR e, lane -> (M = e + 8*lh, N = lr).
#pragma unroll
  for (int tm = 0; tm < 2; ++tm) {
#pragma unroll
    for (int tn = 0; tn < 2; ++tn) {
      const int mbase = m0 + waveM * 32 + tm * 16;
      const int nbase = n0 + waveN * 32 + tn * 16;
      const int ng = nbase + lr;
      float csum = 0.0f;
#pragma unroll
      for (int e = 0; e < 8; ++e) {
        const float ex = __expf(acc[tm][tn][e] * 2.0f);   // 1/tau = 2
        const int mg = mbase + e + 8 * lh;
        if (mg == ng) diag[mg] = ex;                      // unique writer
        csum += ex;
        float rs = ex;                                    // sum over 16 cols
        rs += __shfl_xor(rs, 1, 32);
        rs += __shfl_xor(rs, 2, 32);
        rs += __shfl_xor(rs, 4, 32);
        rs += __shfl_xor(rs, 8, 32);
        if (lr == e)
          atomicAdd(&red_row[waveM * 32 + tm * 16 + e + 8 * lh], rs);
      }
      csum += __shfl_xor(csum, 16, 32);                   // add other M-half
      if (lh == 0)
        atomicAdd(&red_col[waveN * 32 + tn * 16 + lr], csum);
    }
  }
  __syncthreads();
  if (tid < 128)      atomicAdd(&rowsum[m0 + tid], red_row[tid]);
  else if (tid < 192) atomicAdd(&colsum[n0 + (tid - 128)], red_col[tid - 128]);
}

// ------------------------------------------------------------- finalize ---
__global__ __launch_bounds__(256)
void finalize_kernel(const float* __restrict__ rowsum,
                     const float* __restrict__ colsum,
                     const float* __restrict__ diag,
                     float* __restrict__ out, int N) {
  __shared__ float sb[256];
  const int i = blockIdx.x * 256 + threadIdx.x;
  float li = 0.0f;
  if (i < N) {
    const float d = diag[i];
    const float l1 = logf((2.0f * rowsum[i] - d) / d);   // -log(d/denom)
    const float l2 = logf((2.0f * colsum[i] - d) / d);
    li = 0.5f * (l1 + l2);
  }
  sb[threadIdx.x] = li;
  __syncthreads();
  for (int s = 128; s > 0; s >>= 1) {
    if (threadIdx.x < s) sb[threadIdx.x] += sb[threadIdx.x + s];
    __syncthreads();
  }
  if (threadIdx.x == 0) atomicAdd(out, sb[0] / (float)N);
}

// ---------------------------------------------------------------- launch --
extern "C" void kernel_launch(void* const* d_in, const int* in_sizes, int n_in,
                              void* d_out, int out_size, void* d_ws, size_t ws_size,
                              hipStream_t stream) {
  const float* z1 = (const float*)d_in[0];
  const float* z2 = (const float*)d_in[1];
  const int D = 128;
  const int N = in_sizes[0] / D;            // 16384

  char* w = (char*)d_ws;
  const size_t NB = (size_t)N * D * sizeof(unsigned short);   // 4 MB each
  unsigned short* a_hi = (unsigned short*)(w + 0 * NB);
  unsigned short* a_lo = (unsigned short*)(w + 1 * NB);
  unsigned short* b_hi = (unsigned short*)(w + 2 * NB);
  unsigned short* b_lo = (unsigned short*)(w + 3 * NB);
  float* rowsum = (float*)(w + 4 * NB);
  float* colsum = rowsum + N;
  float* diag   = colsum + N;
  float* out    = (float*)d_out;

  init_zero_kernel<<<(3 * N + 255) / 256, 256, 0, stream>>>(rowsum, 3 * N, out);
  normalize_split_kernel<<<(2 * N + 7) / 8, 256, 0, stream>>>(
      z1, z2, a_hi, a_lo, b_hi, b_lo, N);
  dim3 grid(N / 64, N / 128);               // 256 x 128 blocks
  sim_exp_reduce_kernel<<<grid, 256, 0, stream>>>(
      a_hi, a_lo, b_hi, b_lo, rowsum, colsum, diag, N);
  finalize_kernel<<<(N + 255) / 256, 256, 0, stream>>>(rowsum, colsum, diag, out, N);
}
